// RLMamba2Racer_44590350467197
// MI455X (gfx1250) — matrix operand order repaired and checked by reading, not compiled, not run on hardware
//
#include <hip/hip_runtime.h>
#include <hip/hip_bf16.h>
#include <math.h>

// ---------------------------------------------------------------------------
// Model dims
// ---------------------------------------------------------------------------
#define B0     1024
#define LIDAR  1080
#define L1C    538
#define L2C    268
#define L3C    133
#define FLAT   (64 * L3C)       // 8512
#define DM     256
#define DS     16
#define HD     16
#define DIN    512
#define NH     32
#define CONVDIM 544
#define LSEQ   48
#define NROW   (B0 * LSEQ)      // 49152
#define PROJW  (CONVDIM + NH)   // 576 = xBC(544) + dt(32)

typedef __attribute__((ext_vector_type(16))) __bf16 v16bf;
typedef __attribute__((ext_vector_type(8)))  __bf16 v8bf;
typedef __attribute__((ext_vector_type(4)))  __bf16 v4bf;
typedef __attribute__((ext_vector_type(2)))  __bf16 v2bf;
typedef __attribute__((ext_vector_type(8)))  float  v8f;
typedef __attribute__((ext_vector_type(4)))  float  v4f;

// ---------------------------------------------------------------------------
// Generic WMMA GEMM:  C[M,N] = act( A[M,K] @ (W (+W2))[K,N] + bias )
// A row stride sA, W row stride ldw, C row stride sC (elements).
// f32 in/out, bf16 operands, f32 accumulate via v_wmma_f32_16x16x32_bf16.
// Block: 256 threads = 8 waves; tile BM=128 x BN=64, BK=32.
// Wave w: rows 32*(w%4), cols 32*(w/4) -> 2x2 WMMA 16x16 tiles.
// FULL=true requires M%128==0, N%64==0, K%32==0, 16B-aligned A rows:
// register-staged software pipeline + LDS double buffering, one barrier per
// k-step, branch-free inner loop, unconditional k+2 prefetch.
// HASW2 statically selects the dual-weight (W+W2) staging path.
// ---------------------------------------------------------------------------
#define GBM 128
#define GBN 64
#define GBK 32

template <bool FULL, bool HASW2>
__launch_bounds__(256)
__global__ void gemm_bf16_wmma(const float* __restrict__ A, int sA,
                               const float* __restrict__ W,
                               const float* __restrict__ W2, int ldw,
                               const float* __restrict__ bias,
                               float* __restrict__ C, int sC,
                               int M, int N, int K, int act)
{
    __shared__ __align__(16) __bf16 As[2][GBM * GBK];   // [row][k]
    __shared__ __align__(16) __bf16 Bs[2][GBN * GBK];   // [col][k] (K-contig)

    const int tid  = threadIdx.x;
    const int lane = tid & 31;
    const int wave = tid >> 5;
    const int l16  = lane & 15;
    const bool hi  = lane >= 16;

    const int waveM = (wave & 3) * 32;
    const int waveN = (wave >> 2) * 32;
    const int blockM = blockIdx.y * GBM;
    const int blockN = blockIdx.x * GBN;

    v8f acc[2][2];
    #pragma unroll
    for (int i = 0; i < 2; ++i)
        #pragma unroll
        for (int j = 0; j < 2; ++j)
            #pragma unroll
            for (int e = 0; e < 8; ++e) acc[i][j][e] = 0.f;

    // ---- fragment compute from LDS buffer `buf` (2x2 WMMA 16x16) ----
    auto compute = [&](int buf) {
        v16bf afr[2], bfr[2];
        // A frag: lane<16 -> K{0..7,16..23}; lane>=16 -> K{8..15,24..31}
        #pragma unroll
        for (int tm = 0; tm < 2; ++tm) {
            const __bf16* ar = &As[buf][(waveM + tm * 16 + l16) * GBK + (hi ? 8 : 0)];
            v8bf a0 = *(const v8bf*)ar;
            v8bf a1 = *(const v8bf*)(ar + 16);
            afr[tm] = __builtin_shufflevector(a0, a1,
                0,1,2,3,4,5,6,7,8,9,10,11,12,13,14,15);
        }
        // B frag: lane<16 -> K{0..7,16..23}; lane>=16 -> K{8..15,24..31}
        #pragma unroll
        for (int tn = 0; tn < 2; ++tn) {
            const __bf16* br = &Bs[buf][(waveN + tn * 16 + l16) * GBK + (hi ? 8 : 0)];
            v8bf b0 = *(const v8bf*)br;
            v8bf b1 = *(const v8bf*)(br + 16);
            bfr[tn] = __builtin_shufflevector(b0, b1,
                0,1,2,3,4,5,6,7,8,9,10,11,12,13,14,15);
        }
        #pragma unroll
        for (int tm = 0; tm < 2; ++tm)
            #pragma unroll
            for (int tn = 0; tn < 2; ++tn)
                acc[tm][tn] = __builtin_amdgcn_wmma_f32_16x16x32_bf16(
                    false, afr[tm], false, bfr[tn],
                    (short)0, acc[tm][tn], false, false);
    };

    if (FULL) {
        // Per-thread staging addresses (constant across k-steps).
        const float* aptr[4];   int aoff[4];
        const float* wptr[4];   const float* w2ptr[4];   int boff[4];
        #pragma unroll
        for (int it = 0; it < 4; ++it) {
            int v = tid + it * 256;              // 1024 float4 vectors of A
            int r = v >> 3, kc = (v & 7) << 2;   // 8 vectors per 32-wide row
            aptr[it] = A + (long)(blockM + r) * sA + kc;
            aoff[it] = r * GBK + kc;
            int c = v & 63, kp = v >> 6;         // 1024 (col, kpair) units of W
            wptr[it]  = W + (long)(2 * kp) * ldw + blockN + c;
            if (HASW2) w2ptr[it] = W2 + (long)(2 * kp) * ldw + blockN + c;
            boff[it]  = c * GBK + 2 * kp;
        }

        v4f   areg[4];
        float wr0[4], wr1[4];

        // issue 12 global loads (20 with W2) for tile at k0 into registers (+ k+2 prefetch)
        auto gload = [&](int k0) {
            #pragma unroll
            for (int it = 0; it < 4; ++it) {
                const float* ap = aptr[it] + k0;
                areg[it] = *(const v4f*)ap;
                __builtin_prefetch(ap + 2 * GBK, 0, 1);
            }
            #pragma unroll
            for (int it = 0; it < 4; ++it) {
                const float* wp = wptr[it] + (long)k0 * ldw;
                wr0[it] = wp[0]; wr1[it] = wp[ldw];
                __builtin_prefetch(wp + (long)(2 * GBK) * ldw, 0, 1);
                if (HASW2) {
                    const float* wp2 = w2ptr[it] + (long)k0 * ldw;
                    wr0[it] += wp2[0]; wr1[it] += wp2[ldw];
                }
            }
        };
        // convert regs -> LDS buffer `buf`
        auto lstore = [&](int buf) {
            #pragma unroll
            for (int it = 0; it < 4; ++it) {
                v4bf h;
                h[0] = (__bf16)areg[it][0]; h[1] = (__bf16)areg[it][1];
                h[2] = (__bf16)areg[it][2]; h[3] = (__bf16)areg[it][3];
                *(v4bf*)(&As[buf][aoff[it]]) = h;
            }
            #pragma unroll
            for (int it = 0; it < 4; ++it) {
                v2bf h; h[0] = (__bf16)wr0[it]; h[1] = (__bf16)wr1[it];
                *(v2bf*)(&Bs[buf][boff[it]]) = h;
            }
        };

        const int nk = K / GBK;
        gload(0);
        lstore(0);
        __syncthreads();
        for (int kt = 0; kt < nk - 1; ++kt) {        // branch-free pipeline
            gload((kt + 1) * GBK);                   // regs <- tile kt+1
            compute(kt & 1);                         // wmma  <- LDS[cur]
            lstore((kt + 1) & 1);                    // LDS[nxt] <- regs
            __syncthreads();
        }
        compute((nk - 1) & 1);                       // epilogue
    } else {
        // ---- guarded scalar staging (ragged dims; edge layers only) ----
        for (int k0 = 0; k0 < K; k0 += GBK) {
            #pragma unroll
            for (int it = 0; it < 16; ++it) {
                int idx = tid + it * 256;
                int r = idx >> 5, kk = idx & 31;
                int gr = blockM + r, gk = k0 + kk;
                float v = (gr < M && gk < K) ? A[(long)gr * sA + gk] : 0.f;
                As[0][r * GBK + kk] = (__bf16)v;
            }
            #pragma unroll
            for (int it = 0; it < 8; ++it) {
                int idx = tid + it * 256;
                int c = idx & 63, kk = idx >> 6;
                int gc = blockN + c, gk = k0 + kk;
                float v = 0.f;
                if (gc < N && gk < K) {
                    v = W[(long)gk * ldw + gc];
                    if (HASW2) v += W2[(long)gk * ldw + gc];
                }
                Bs[0][c * GBK + kk] = (__bf16)v;
            }
            __syncthreads();
            compute(0);
            __syncthreads();
        }
    }

    // C layout: VGPR r -> M = r (+8 for upper half-wave); N = lane%16.
    #pragma unroll
    for (int tm = 0; tm < 2; ++tm)
        #pragma unroll
        for (int tn = 0; tn < 2; ++tn) {
            int rbase = blockM + waveM + tm * 16 + (hi ? 8 : 0);
            int gc = blockN + waveN + tn * 16 + l16;
            if (!FULL && gc >= N) continue;
            float bv = bias ? bias[gc] : 0.f;
            #pragma unroll
            for (int r = 0; r < 8; ++r) {
                int gr = rbase + r;
                if (FULL || gr < M) {
                    float v = acc[tm][tn][r] + bv;
                    if (act == 1) v = fmaxf(v, 0.f);
                    C[(long)gr * sC + gc] = v;
                }
            }
        }
}

// ---------------------------------------------------------------------------
// Fused CNN stack: one block per batch element, ping-pong in dynamic LDS.
// ---------------------------------------------------------------------------
__launch_bounds__(256)
__global__ void conv_stack_kernel(const float* __restrict__ scan,
                                  const float* __restrict__ cw1, const float* __restrict__ cb1,
                                  const float* __restrict__ cw2, const float* __restrict__ cb2,
                                  const float* __restrict__ cw3, const float* __restrict__ cb3,
                                  float* __restrict__ h3)
{
    extern __shared__ float smem[];
    float* bufA = smem;                 // 32*538 = 17216 (conv1 out)
    float* bufB = smem + 32 * L1C;      // 17152 (input, then conv2 out)
    const int b = blockIdx.x;
    const int tid = threadIdx.x;

    for (int i = tid; i < LIDAR; i += 256) bufB[i] = scan[(long)b * LIDAR + i];
    __syncthreads();

    for (int i = tid; i < 32 * L1C; i += 256) {         // k=5, stride 2
        int c = i / L1C, t = i % L1C;
        float a = cb1[c];
        const float* x = &bufB[2 * t];
        #pragma unroll
        for (int j = 0; j < 5; ++j) a += cw1[c * 5 + j] * x[j];
        bufA[c * L1C + t] = fmaxf(a, 0.f);
    }
    __syncthreads();

    for (int i = tid; i < 64 * L2C; i += 256) {         // 32ch, k=3, stride 2
        int c = i / L2C, t = i % L2C;
        float a = cb2[c];
        for (int ci = 0; ci < 32; ++ci) {
            const float* w = &cw2[(c * 32 + ci) * 3];
            const float* x = &bufA[ci * L1C + 2 * t];
            a += w[0] * x[0] + w[1] * x[1] + w[2] * x[2];
        }
        bufB[c * L2C + t] = fmaxf(a, 0.f);
    }
    __syncthreads();

    for (int i = tid; i < 64 * L3C; i += 256) {         // 64ch, k=3, stride 2
        int c = i / L3C, t = i % L3C;
        float a = cb3[c];
        for (int ci = 0; ci < 64; ++ci) {
            const float* w = &cw3[(c * 64 + ci) * 3];
            const float* x = &bufB[ci * L2C + 2 * t];
            a += w[0] * x[0] + w[1] * x[1] + w[2] * x[2];
        }
        h3[(long)b * FLAT + c * L3C + t] = fmaxf(a, 0.f);   // reshape(b,-1) order
    }
}

// ---------------------------------------------------------------------------
// Odometry projection -> comb[:, 256:320]
// ---------------------------------------------------------------------------
__global__ void odom_kernel(const float* __restrict__ state,
                            const float* __restrict__ ow, const float* __restrict__ obv,
                            const int* __restrict__ maxspd, float* __restrict__ comb)
{
    int i = blockIdx.x * blockDim.x + threadIdx.x;
    if (i >= B0 * 64) return;
    int b = i >> 6, o = i & 63;
    float inv = 1.f / (float)maxspd[0];
    float a = obv[o];
    #pragma unroll
    for (int s = 0; s < 4; ++s) a += (state[b * 4 + s] * inv) * ow[s * 64 + o];
    comb[(long)b * 320 + 256 + o] = a;
}

// ---------------------------------------------------------------------------
// Buffer shift + LayerNorm + clip -> writes buf output and xn.
// ---------------------------------------------------------------------------
__launch_bounds__(256)
__global__ void bufln_kernel(const float* __restrict__ obs, const float* __restrict__ f2,
                             const float* __restrict__ g, const float* __restrict__ beta,
                             float* __restrict__ bufout, float* __restrict__ xn)
{
    int row = blockIdx.x;                  // b*48 + l
    int b = row / LSEQ, l = row % LSEQ;
    int tid = threadIdx.x;
    const float* src = (l < LSEQ - 1) ? &obs[((long)b * LSEQ + l + 1) * DM]
                                      : &f2[(long)b * DM];
    float v = src[tid];
    bufout[(long)row * DM + tid] = v;

    __shared__ float red[256];
    red[tid] = v; __syncthreads();
    for (int s = 128; s > 0; s >>= 1) { if (tid < s) red[tid] += red[tid + s]; __syncthreads(); }
    float mean = red[0] * (1.f / DM);
    __syncthreads();
    float d = v - mean;
    red[tid] = d * d; __syncthreads();
    for (int s = 128; s > 0; s >>= 1) { if (tid < s) red[tid] += red[tid + s]; __syncthreads(); }
    float var = red[0] * (1.f / DM);
    float y = d * rsqrtf(var + 1e-5f) * g[tid] + beta[tid];
    xn[(long)row * DM + tid] = fminf(fmaxf(y, -10.f), 10.f);
}

// ---------------------------------------------------------------------------
// Post in_proj: dt softplus + dA, causal depthwise conv (k=4) + SiLU in place.
// ---------------------------------------------------------------------------
__global__ void postproj_kernel(float* __restrict__ proj,
                                const float* __restrict__ conv_w, const float* __restrict__ conv_b,
                                const float* __restrict__ A_log, const float* __restrict__ dt_bias,
                                float* __restrict__ dAbuf)
{
    int i = blockIdx.x * blockDim.x + threadIdx.x;
    if (i >= B0 * PROJW) return;
    int b = i / PROJW, c = i % PROJW;
    if (c < CONVDIM) {
        float w0 = conv_w[c * 4 + 0], w1 = conv_w[c * 4 + 1];
        float w2 = conv_w[c * 4 + 2], w3 = conv_w[c * 4 + 3];
        float bb = conv_b[c];
        float x0 = 0.f, x1 = 0.f, x2 = 0.f, x3 = 0.f;
        for (int t = 0; t < LSEQ; ++t) {
            long idx = ((long)b * LSEQ + t) * PROJW + c;
            x0 = x1; x1 = x2; x2 = x3; x3 = proj[idx];
            float o = bb + w0 * x0 + w1 * x1 + w2 * x2 + w3 * x3;
            float sg = 1.f / (1.f + expf(-o));
            proj[idx] = o * sg;                         // SiLU
        }
    } else {
        int h = c - CONVDIM;
        float A = -expf(A_log[h]);
        float db = dt_bias[h];
        for (int t = 0; t < LSEQ; ++t) {
            long row = (long)b * LSEQ + t;
            long idx = row * PROJW + CONVDIM + h;
            float dv = proj[idx] + db;
            float sp = (dv > 20.f) ? dv : log1pf(expf(dv));
            proj[idx] = sp;                             // softplus(dt)
            dAbuf[row * NH + h] = expf(sp * A);
        }
    }
}

// ---------------------------------------------------------------------------
// Mamba2 selective scan. One block per (b,h): 256 threads = (p,n) state grid.
// Only y at t = L-1 is emitted (only the last timestep is consumed).
// ---------------------------------------------------------------------------
__launch_bounds__(256)
__global__ void scan_kernel(const float* __restrict__ proj, const float* __restrict__ dAbuf,
                            const float* __restrict__ Dp, float* __restrict__ ylast)
{
    int bh = blockIdx.x;
    int b = bh >> 5, h = bh & 31;
    int tid = threadIdx.x;
    int p = tid >> 4, n = tid & 15;
    float s = 0.f;
    for (int t = 0; t < LSEQ; ++t) {
        long row = (long)b * LSEQ + t;
        const float* pr = &proj[row * PROJW];
        float x  = pr[h * HD + p];
        float Bv = pr[DIN + n];
        float Cv = pr[DIN + DS + n];
        float dt = pr[CONVDIM + h];
        float dA = dAbuf[row * NH + h];
        s = s * dA + dt * x * Bv;
        if (t == LSEQ - 1) {
            float v = s * Cv;
            v += __shfl_xor(v, 1, 16);
            v += __shfl_xor(v, 2, 16);
            v += __shfl_xor(v, 4, 16);
            v += __shfl_xor(v, 8, 16);
            if (n == 0) ylast[(long)b * DIN + h * HD + p] = v + Dp[h] * x;
        }
    }
}

// ---------------------------------------------------------------------------
// Gated RMSNorm (last step only): y = rmsnorm(y * silu(z)) * rms_g, in place.
// ---------------------------------------------------------------------------
__launch_bounds__(256)
__global__ void gaterms_kernel(float* __restrict__ y, const float* __restrict__ z,
                               const float* __restrict__ rg)
{
    int b = blockIdx.x, tid = threadIdx.x;
    __shared__ float red[256];
    float vals[2]; float acc = 0.f;
    #pragma unroll
    for (int j = 0; j < 2; ++j) {
        int c = tid + j * 256;
        float zv = z[(long)b * DIN + c];
        float g = zv / (1.f + expf(-zv));
        float v = y[(long)b * DIN + c] * g;
        vals[j] = v; acc += v * v;
    }
    red[tid] = acc; __syncthreads();
    for (int s = 128; s > 0; s >>= 1) { if (tid < s) red[tid] += red[tid + s]; __syncthreads(); }
    float scale = rsqrtf(red[0] * (1.f / DIN) + 1e-5f);
    #pragma unroll
    for (int j = 0; j < 2; ++j) {
        int c = tid + j * 256;
        y[(long)b * DIN + c] = vals[j] * scale * rg[c];
    }
}

// ---------------------------------------------------------------------------
// LayerNorm (256 wide) -> out with row stride ostride.
// ---------------------------------------------------------------------------
__launch_bounds__(256)
__global__ void ln_kernel(const float* __restrict__ x, const float* __restrict__ g,
                          const float* __restrict__ beta, float* __restrict__ out, int ostride)
{
    int b = blockIdx.x, tid = threadIdx.x;
    __shared__ float red[256];
    float v = x[(long)b * DM + tid];
    red[tid] = v; __syncthreads();
    for (int s = 128; s > 0; s >>= 1) { if (tid < s) red[tid] += red[tid + s]; __syncthreads(); }
    float mean = red[0] * (1.f / DM);
    __syncthreads();
    float d = v - mean;
    red[tid] = d * d; __syncthreads();
    for (int s = 128; s > 0; s >>= 1) { if (tid < s) red[tid] += red[tid + s]; __syncthreads(); }
    float var = red[0] * (1.f / DM);
    out[(long)b * ostride + tid] = d * rsqrtf(var + 1e-5f) * g[tid] + beta[tid];
}

// ---------------------------------------------------------------------------
// Output heads: loc = clip(x @ mw + mb), scale = exp(clip(log_std)).
// ---------------------------------------------------------------------------
__global__ void heads_kernel(const float* __restrict__ x, const float* __restrict__ mw,
                             const float* __restrict__ mb, const float* __restrict__ log_std,
                             float* __restrict__ out)
{
    int i = blockIdx.x * blockDim.x + threadIdx.x;
    if (i >= B0 * 2) return;
    int b = i >> 1, j = i & 1;
    float a = mb[j];
    for (int k = 0; k < 32; ++k) a += x[(long)b * 512 + k] * mw[k * 2 + j];
    out[b * 2 + j] = fminf(fmaxf(a, -10.f), 10.f);
    float ls = fminf(fmaxf(log_std[j], -2.f), 2.f);
    out[B0 * 2 + b * 2 + j] = expf(ls);
}

// ---------------------------------------------------------------------------
// Host-side orchestration
// ---------------------------------------------------------------------------
extern "C" void kernel_launch(void* const* d_in, const int* in_sizes, int n_in,
                              void* d_out, int out_size, void* d_ws, size_t ws_size,
                              hipStream_t stream)
{
    (void)in_sizes; (void)n_in; (void)out_size; (void)ws_size;

    const float* scan   = (const float*)d_in[0];
    const float* state  = (const float*)d_in[1];
    const float* obs    = (const float*)d_in[2];
    const float* cw1 = (const float*)d_in[3];  const float* cb1 = (const float*)d_in[4];
    const float* cw2 = (const float*)d_in[5];  const float* cb2 = (const float*)d_in[6];
    const float* cw3 = (const float*)d_in[7];  const float* cb3 = (const float*)d_in[8];
    const float* vw  = (const float*)d_in[9];  const float* vb  = (const float*)d_in[10];
    const float* ow  = (const float*)d_in[11]; const float* obv = (const float*)d_in[12];
    const float* f1w = (const float*)d_in[13]; const float* f1b = (const float*)d_in[14];
    const float* f2w = (const float*)d_in[15]; const float* f2b = (const float*)d_in[16];
    const float* pn_g = (const float*)d_in[17]; const float* pn_b = (const float*)d_in[18];
    const float* in_w = (const float*)d_in[19];
    const float* conv_w = (const float*)d_in[20]; const float* conv_b = (const float*)d_in[21];
    const float* A_log = (const float*)d_in[22]; const float* dt_bias = (const float*)d_in[23];
    const float* Dp = (const float*)d_in[24];   const float* rms_g = (const float*)d_in[25];
    const float* out_w = (const float*)d_in[26];
    const float* ln_g = (const float*)d_in[27]; const float* ln_b = (const float*)d_in[28];
    // res[i]: (W,b,P) at indices 29+3i, 30+3i, 31+3i
    const float* lw = (const float*)d_in[47]; const float* lb = (const float*)d_in[48];
    const float* mw = (const float*)d_in[49]; const float* mb = (const float*)d_in[50];
    const float* log_std = (const float*)d_in[51];
    const int*   maxspd  = (const int*)d_in[52];

    float* out = (float*)d_out;
    float* ws  = (float*)d_ws;

    // Workspace layout (element offsets). proj region reuses CNN/MLP scratch.
    float* proj = ws + 0;                   // 49152 x 576
    float* h3   = ws + 0;                   // 1024 x 8512
    float* comb = ws + 8716288;             // 1024 x 320
    float* f1o  = ws + 9043968;             // 1024 x 768
    float* f2o  = ws + 9830400;             // 1024 x 256
    float* xn   = ws + 28311552;            // 49152 x 256
    float* dA   = ws + 40894464;            // 49152 x 32
    float* zl   = ws + 42467328;            // 1024 x 512  (z at last step)
    float* yl   = ws + 42991616;            // 1024 x 512  (scan y at last step)
    float* mbuf = ws + 43515904;            // 1024 x 256
    float* resA = ws + 43778048;            // 1024 x 512 ping
    float* resB = ws + 44302336;            // 1024 x 512 pong

    // 1. CNN stack (fused, LDS ping-pong)
    conv_stack_kernel<<<B0, 256, (32 * L1C + 64 * L2C) * sizeof(float), stream>>>(
        scan, cw1, cb1, cw2, cb2, cw3, cb3, h3);

    // 2. odometry -> comb[:,256:]
    odom_kernel<<<(B0 * 64 + 255) / 256, 256, 0, stream>>>(state, ow, obv, maxspd, comb);

    // 3. vis = relu(h3 @ vw + vb) -> comb[:,0:256]      (FULL: 256|64, 8512|32)
    gemm_bf16_wmma<true, false><<<dim3(DM / GBN, B0 / GBM), 256, 0, stream>>>(
        h3, FLAT, vw, nullptr, DM, vb, comb, 320, B0, DM, FLAT, 1);

    // 4. f1 = relu(comb @ f1w + f1b)                    (FULL: 768|64, 320|32)
    gemm_bf16_wmma<true, false><<<dim3(768 / GBN, B0 / GBM), 256, 0, stream>>>(
        comb, 320, f1w, nullptr, 768, f1b, f1o, 768, B0, 768, 320, 1);

    // 5. f2 = relu(f1 @ f2w + f2b)                      (FULL)
    gemm_bf16_wmma<true, false><<<dim3(DM / GBN, B0 / GBM), 256, 0, stream>>>(
        f1o, 768, f2w, nullptr, DM, f2b, f2o, DM, B0, DM, 768, 1);

    // 6. buffer shift + LN + clip -> buf output (d_out+4096) and xn
    bufln_kernel<<<NROW, 256, 0, stream>>>(obs, f2o, pn_g, pn_b, out + 2 * B0 * 2, xn);

    // 7a. in_proj, xBC+dt columns (512..1087) for all rows   (FULL: 576|64, 256|32)
    gemm_bf16_wmma<true, false><<<dim3(PROJW / GBN, NROW / GBM), 256, 0, stream>>>(
        xn, DM, in_w + DIN, nullptr, 1088, nullptr, proj, PROJW, NROW, PROJW, DM, 0);

    // 7b. z columns (0..511), last timestep rows only (strided A)  (FULL)
    gemm_bf16_wmma<true, false><<<dim3(DIN / GBN, B0 / GBM), 256, 0, stream>>>(
        xn + (LSEQ - 1) * DM, LSEQ * DM, in_w, nullptr, 1088, nullptr, zl, DIN,
        B0, DIN, DM, 0);

    // 8. softplus(dt)+dA, causal depthwise conv + SiLU (in place)
    postproj_kernel<<<(B0 * PROJW + 255) / 256, 256, 0, stream>>>(
        proj, conv_w, conv_b, A_log, dt_bias, dA);

    // 9. selective scan -> y at last step (+ D*x)
    scan_kernel<<<B0 * NH, 256, 0, stream>>>(proj, dA, Dp, yl);

    // 10. gated RMSNorm
    gaterms_kernel<<<B0, 256, 0, stream>>>(yl, zl, rms_g);

    // 11. out_proj: m = y @ out_w                        (FULL: 256|64, 512|32)
    gemm_bf16_wmma<true, false><<<dim3(DM / GBN, B0 / GBM), 256, 0, stream>>>(
        yl, DIN, out_w, nullptr, DM, nullptr, mbuf, DM, B0, DM, DIN, 0);

    // 12. final LN -> resA
    ln_kernel<<<B0, 256, 0, stream>>>(mbuf, ln_g, ln_b, resA, 512);

    // 13. residual stack: relu(x @ (W+P) + b), 6 layers, ping-pong
    const int dims[7] = {256, 496, 382, 296, 192, 128, 64};
    float* bufs[2] = {resA, resB};
    for (int i = 0; i < 6; ++i) {
        const float* Wp = (const float*)d_in[29 + 3 * i];
        const float* bp = (const float*)d_in[30 + 3 * i];
        const float* Pp = (const float*)d_in[31 + 3 * i];
        bool full = (dims[i + 1] % GBN == 0) && (dims[i] % GBK == 0);
        dim3 gg((dims[i + 1] + GBN - 1) / GBN, B0 / GBM);
        if (full)
            gemm_bf16_wmma<true, true><<<gg, 256, 0, stream>>>(
                bufs[i & 1], 512, Wp, Pp, dims[i + 1], bp, bufs[(i + 1) & 1], 512,
                B0, dims[i + 1], dims[i], 1);
        else
            gemm_bf16_wmma<false, true><<<gg, 256, 0, stream>>>(
                bufs[i & 1], 512, Wp, Pp, dims[i + 1], bp, bufs[(i + 1) & 1], 512,
                B0, dims[i + 1], dims[i], 1);
    }
    // output of layer 5 is in resA (64 wide)

    // 14. x = relu(x @ lw + lb) -> resB (32 wide, ragged N -> guarded)
    gemm_bf16_wmma<false, false><<<dim3(1, B0 / GBM), 256, 0, stream>>>(
        resA, 512, lw, nullptr, 32, lb, resB, 512, B0, 32, 64, 1);

    // 15. heads: loc + scale
    heads_kernel<<<(B0 * 2 + 255) / 256, 256, 0, stream>>>(resB, mw, mb, log_std, out);
}
